// MaskedMultiHeadAttentionBlock_28905129902444
// MI455X (gfx1250) — matrix operand order repaired and checked by reading, not compiled, not run on hardware
//
#include <hip/hip_runtime.h>
#include <math.h>

// Problem constants (from reference): B=2, N=2048, H=1024, NH=16, HD=64
#define B_   2
#define N_   2048
#define H_   1024
#define NH_  16
#define HD_  64
#define R_   (B_ * N_)          // 4096 rows for the projection GEMMs
#define BNH_ ((size_t)B_ * N_ * H_)

typedef __attribute__((ext_vector_type(16))) _Float16 v16h;
typedef __attribute__((ext_vector_type(8)))  _Float16 v8h;
typedef __attribute__((ext_vector_type(8)))  float    v8f;

union Frag16 { v16h v; v8h h[2]; };

__device__ __forceinline__ v8f wmma_f16(const Frag16& a, const Frag16& b, v8f c) {
    // D = A(16x32 f16) * B(32x16 f16) + C(16x16 f32)   -> v_wmma_f32_16x16x32_f16
    return __builtin_amdgcn_wmma_f32_16x16x32_f16(
        /*neg_a=*/false, a.v, /*neg_b=*/false, b.v,
        /*c_mod=*/(short)0, c, /*reuse_a=*/false, /*reuse_b=*/false);
}

__device__ __forceinline__ v8h cvt8(const float4 a, const float4 b) {
    v8h r = { (_Float16)a.x, (_Float16)a.y, (_Float16)a.z, (_Float16)a.w,
              (_Float16)b.x, (_Float16)b.y, (_Float16)b.z, (_Float16)b.w };
    return r;
}

// gfx1250 async global->LDS copy (16B per lane), tracked by ASYNCcnt.
__device__ __forceinline__ void async_copy16(void* lds_dst, const void* gsrc) {
    const unsigned  ldsoff = (unsigned)(uintptr_t)lds_dst;        // lo32 of shared aperture = LDS offset
    const unsigned long long ga = (unsigned long long)(uintptr_t)gsrc;
    asm volatile("global_load_async_to_lds_b128 %0, %1, off"
                 :: "v"(ldsoff), "v"(ga) : "memory");
}
__device__ __forceinline__ void wait_async0() {
    asm volatile("s_wait_asynccnt 0x0" ::: "memory");
}

// ---------------------------------------------------------------------------
// Projection GEMM:  Y[r, i] = sum_j X[r, j] * W[i, j] + bias[i]
// X: [R_, H_] fp32, W: [H_, H_] fp32, Y: [R_, H_] (f16 for Q/K/V, f32 for final)
// Block: 256 threads (8 waves), tile 128x128, K-step 32.
// ---------------------------------------------------------------------------
#define TS 40   // LDS row stride in halfs (32 data + 8 pad; 80B, 16B aligned)

template <typename Tout>
__global__ __launch_bounds__(256)
void proj_gemm(const float* __restrict__ X, const float* __restrict__ W,
               const float* __restrict__ bias, Tout* __restrict__ Y)
{
    __shared__ _Float16 Xs[128 * TS];
    __shared__ _Float16 Ws[128 * TS];

    const int tid  = threadIdx.x;
    const int lane = tid & 31, wid = tid >> 5;
    const int h = lane >> 4, r15 = lane & 15;
    const int mblk = blockIdx.x * 128;        // output row block
    const int nblk = blockIdx.y * 128;        // output col block
    const int mw = (wid >> 2) * 64;           // wave row offset inside tile
    const int nw = (wid & 3) * 32;            // wave col offset inside tile

    v8f acc[4][2];
    #pragma unroll
    for (int mt = 0; mt < 4; ++mt)
        #pragma unroll
        for (int nt = 0; nt < 2; ++nt)
            acc[mt][nt] = (v8f){0.f,0.f,0.f,0.f,0.f,0.f,0.f,0.f};

    const int sr = tid >> 1;          // staging row 0..127
    const int sk = (tid & 1) * 16;    // staging k offset 0/16

    for (int kt = 0; kt < H_; kt += 32) {
        __syncthreads();
        // ---- stage X and W tiles (fp32 -> f16 conversion in flight) ----
        {
            const float4* src = (const float4*)(X + (size_t)(mblk + sr) * H_ + kt + sk);
            float4 f0 = src[0], f1 = src[1], f2 = src[2], f3 = src[3];
            *(v8h*)&Xs[sr * TS + sk]     = cvt8(f0, f1);
            *(v8h*)&Xs[sr * TS + sk + 8] = cvt8(f2, f3);
            const float4* wsrc = (const float4*)(W + (size_t)(nblk + sr) * H_ + kt + sk);
            float4 g0 = wsrc[0], g1 = wsrc[1], g2 = wsrc[2], g3 = wsrc[3];
            *(v8h*)&Ws[sr * TS + sk]     = cvt8(g0, g1);
            *(v8h*)&Ws[sr * TS + sk + 8] = cvt8(g2, g3);
        }
        __syncthreads();

        // ---- fragments per ISA 7.12.2 layouts ----
        Frag16 A[4], Bf[2];
        #pragma unroll
        for (int mt = 0; mt < 4; ++mt) {
            const _Float16* p = &Xs[(mw + mt * 16 + r15) * TS];
            A[mt].h[0] = *(const v8h*)(p + h * 8);          // K = h*8 + 0..7
            A[mt].h[1] = *(const v8h*)(p + 16 + h * 8);     // K = 16 + h*8 + 0..7
        }
        #pragma unroll
        for (int nt = 0; nt < 2; ++nt) {
            const _Float16* p = &Ws[(nw + nt * 16 + r15) * TS];
            Bf[nt].h[0] = *(const v8h*)(p + h * 8);         // K = h*8 + 0..7
            Bf[nt].h[1] = *(const v8h*)(p + 16 + h * 8);    // K = 16 + h*8 + 0..7
        }
        #pragma unroll
        for (int mt = 0; mt < 4; ++mt)
            #pragma unroll
            for (int nt = 0; nt < 2; ++nt)
                acc[mt][nt] = wmma_f16(A[mt], Bf[nt], acc[mt][nt]);
    }

    // ---- epilogue: bias + store (C layout: lane = col, VGPR v = row h*8+v) ----
    #pragma unroll
    for (int nt = 0; nt < 2; ++nt) {
        const int col = nblk + nw + nt * 16 + r15;
        const float bv = bias[col];
        #pragma unroll
        for (int mt = 0; mt < 4; ++mt) {
            #pragma unroll
            for (int v = 0; v < 8; ++v) {
                const int row = mblk + mw + mt * 16 + h * 8 + v;
                Y[(size_t)row * H_ + col] = (Tout)(acc[mt][nt][v] + bv);
            }
        }
    }
}

// ---------------------------------------------------------------------------
// Flash attention (causal). Q/K/V: f16 [B, N, H] (head-major inside H).
// Block: 256 threads = 8 waves; each wave owns a 16x64 Q tile (128 rows/block).
// K tile staged via async global->LDS (no VGPR round trip); V staged with a
// register transpose; softmax denominator computed with a ones-WMMA.
// ---------------------------------------------------------------------------
#define KST 72   // K_lds row stride (64 + 8 pad)
#define VST 40   // V_lds row stride (32 + 8 pad)
#define PST 40   // P_lds row stride (32 + 8 pad)

__global__ __launch_bounds__(256)
void attn_kernel(const _Float16* __restrict__ Q, const _Float16* __restrict__ K,
                 const _Float16* __restrict__ V, float* __restrict__ O)
{
    __shared__ _Float16 Ks[32 * KST];
    __shared__ _Float16 Vs[64 * VST];          // transposed: [d][m]
    __shared__ _Float16 Ps[8 * 16 * PST];      // per-wave P scratch

    const int tid = threadIdx.x, lane = tid & 31, wid = tid >> 5;
    const int h = lane >> 4, r15 = lane & 15;
    const int bh = blockIdx.y;
    const int b = bh / NH_, head = bh % NH_;
    const int q0 = blockIdx.x * 128;
    const int qrow0 = q0 + wid * 16;

    const size_t base = (size_t)b * N_ * H_ + head * HD_;

    // Q A-fragments: straight from global (d contiguous -> 16B loads), kept live.
    Frag16 Qf[2];
    {
        const _Float16* qp = Q + base + (size_t)(qrow0 + r15) * H_;
        #pragma unroll
        for (int dh = 0; dh < 2; ++dh) {
            Qf[dh].h[0] = *(const v8h*)(qp + dh * 32 + h * 8);
            Qf[dh].h[1] = *(const v8h*)(qp + dh * 32 + 16 + h * 8);
        }
    }

    // Constant all-ones B fragment: rowsum(P) = P x Ones via one WMMA.
    Frag16 OnesF;
    #pragma unroll
    for (int i = 0; i < 16; ++i) OnesF.v[i] = (_Float16)1.0f;

    v8f o[4];
    #pragma unroll
    for (int dt = 0; dt < 4; ++dt) o[dt] = (v8f){0.f,0.f,0.f,0.f,0.f,0.f,0.f,0.f};
    float mrun[8], srun[8];
    #pragma unroll
    for (int v = 0; v < 8; ++v) { mrun[v] = -__builtin_inff(); srun[v] = 0.f; }

    const int sm = tid >> 3;          // staging row 0..31
    const int sd = (tid & 7) * 8;     // staging d 0..56

    const int jend = q0 + 128;        // causal: max col needed = q0+127
    for (int jb = 0; jb < jend; jb += 32) {
        __syncthreads();
        // ---- stage K (async DMA to LDS) and transposed V ----
        {
            const size_t rofs = base + (size_t)(jb + sm) * H_ + sd;
            async_copy16(&Ks[sm * KST + sd], K + rofs);       // ASYNCcnt path
            const v8h vv = *(const v8h*)(V + rofs);
            #pragma unroll
            for (int q = 0; q < 8; ++q) Vs[(sd + q) * VST + sm] = vv[q];
            wait_async0();
        }
        __syncthreads();

        // ---- S = Q * K^T  (two 16x16 tiles, contraction d=64 in 2 halves) ----
        v8f s0 = (v8f){0.f,0.f,0.f,0.f,0.f,0.f,0.f,0.f};
        v8f s1 = (v8f){0.f,0.f,0.f,0.f,0.f,0.f,0.f,0.f};
        #pragma unroll
        for (int dh = 0; dh < 2; ++dh) {
            Frag16 Kf;
            const _Float16* kb0 = &Ks[r15 * KST + dh * 32];
            Kf.h[0] = *(const v8h*)(kb0 + h * 8); Kf.h[1] = *(const v8h*)(kb0 + 16 + h * 8);
            s0 = wmma_f16(Qf[dh], Kf, s0);
            const _Float16* kb1 = &Ks[(16 + r15) * KST + dh * 32];
            Kf.h[0] = *(const v8h*)(kb1 + h * 8); Kf.h[1] = *(const v8h*)(kb1 + 16 + h * 8);
            s1 = wmma_f16(Qf[dh], Kf, s1);
        }

        // ---- masked, scaled scores; batched row-max butterfly ----
        const int c0 = jb + r15, c1 = c0 + 16;
        float sa[8], sb[8], tv[8];
        #pragma unroll
        for (int v = 0; v < 8; ++v) {
            const int row = qrow0 + h * 8 + v;
            sa[v] = (c0 <= row) ? s0[v] * 0.125f : -__builtin_inff();
            sb[v] = (c1 <= row) ? s1[v] * 0.125f : -__builtin_inff();
            tv[v] = fmaxf(sa[v], sb[v]);
        }
        // stage-wise: 8 independent bpermutes per stage -> latencies overlap
        #pragma unroll
        for (int off = 1; off < 16; off <<= 1) {
            float rv[8];
            #pragma unroll
            for (int v = 0; v < 8; ++v) rv[v] = __shfl_xor(tv[v], off, 32);
            #pragma unroll
            for (int v = 0; v < 8; ++v) tv[v] = fmaxf(tv[v], rv[v]);
        }

        float alpha[8], p0a[8], p1a[8];
        #pragma unroll
        for (int v = 0; v < 8; ++v) {
            const float mn = fmaxf(mrun[v], tv[v]);
            alpha[v] = __expf(mrun[v] - mn);
            mrun[v] = mn;
            p0a[v] = __expf(sa[v] - mn);
            p1a[v] = __expf(sb[v] - mn);
            #pragma unroll
            for (int dt = 0; dt < 4; ++dt) o[dt][v] *= alpha[v];
        }

        // C-layout -> A-frag layout via per-wave LDS round trip
        _Float16* prow = &Ps[wid * 16 * PST];
        #pragma unroll
        for (int v = 0; v < 8; ++v) {
            const int i = h * 8 + v;
            prow[i * PST + r15]      = (_Float16)p0a[v];
            prow[i * PST + 16 + r15] = (_Float16)p1a[v];
        }
        asm volatile("s_wait_dscnt 0x0" ::: "memory");   // same-wave LDS RAW

        Frag16 Pf;
        {
            const _Float16* pp = &prow[r15 * PST];
            Pf.h[0] = *(const v8h*)(pp + h * 8);
            Pf.h[1] = *(const v8h*)(pp + 16 + h * 8);
        }

        // ---- softmax denominator: rowsum(P) via ones-WMMA (no shuffles) ----
        {
            v8f sumt = wmma_f16(Pf, OnesF,
                                (v8f){0.f,0.f,0.f,0.f,0.f,0.f,0.f,0.f});
            #pragma unroll
            for (int v = 0; v < 8; ++v) srun[v] = srun[v] * alpha[v] + sumt[v];
        }

        // ---- O += P(16x32) * V(32x64) : one WMMA per 16-wide d sub-tile ----
        #pragma unroll
        for (int dt = 0; dt < 4; ++dt) {
            Frag16 Vf;
            const _Float16* vp = &Vs[(dt * 16 + r15) * VST];
            Vf.h[0] = *(const v8h*)(vp + h * 8); Vf.h[1] = *(const v8h*)(vp + 16 + h * 8);
            o[dt] = wmma_f16(Pf, Vf, o[dt]);
        }
    }

    // ---- normalize and write attention output (merged-head layout) ----
    #pragma unroll
    for (int v = 0; v < 8; ++v) {
        const float inv = 1.0f / srun[v];
        const int row = qrow0 + h * 8 + v;
        float* op = O + base + (size_t)row * H_;
        #pragma unroll
        for (int dt = 0; dt < 4; ++dt)
            op[dt * 16 + r15] = o[dt][v] * inv;
    }
}

// ---------------------------------------------------------------------------
// Launch: 3 projection GEMMs -> flash attention -> output projection.
// d_in order: q,k,v,mask,Wq,bq,Wk,bk,Wv,bv,Wp,bp  (mask is causal tril; we
// apply causality by index compare and never read it).
// ---------------------------------------------------------------------------
extern "C" void kernel_launch(void* const* d_in, const int* in_sizes, int n_in,
                              void* d_out, int out_size, void* d_ws, size_t ws_size,
                              hipStream_t stream)
{
    const float* q  = (const float*)d_in[0];
    const float* k  = (const float*)d_in[1];
    const float* v  = (const float*)d_in[2];
    // d_in[3] = mask (unused; causal handled analytically)
    const float* Wq = (const float*)d_in[4];
    const float* bq = (const float*)d_in[5];
    const float* Wk = (const float*)d_in[6];
    const float* bk = (const float*)d_in[7];
    const float* Wv = (const float*)d_in[8];
    const float* bv = (const float*)d_in[9];
    const float* Wp = (const float*)d_in[10];
    const float* bp = (const float*)d_in[11];

    char* ws = (char*)d_ws;
    _Float16* Qb = (_Float16*)(ws);
    _Float16* Kb = (_Float16*)(ws + BNH_ * 2);
    _Float16* Vb = (_Float16*)(ws + BNH_ * 4);
    float*    AO = (float*)   (ws + BNH_ * 6);

    dim3 pgrid(R_ / 128, H_ / 128);   // 32 x 8
    dim3 pblk(256);
    proj_gemm<_Float16><<<pgrid, pblk, 0, stream>>>(q, Wq, bq, Qb);
    proj_gemm<_Float16><<<pgrid, pblk, 0, stream>>>(k, Wk, bk, Kb);
    proj_gemm<_Float16><<<pgrid, pblk, 0, stream>>>(v, Wv, bv, Vb);

    dim3 agrid(N_ / 128, B_ * NH_);   // 16 x 32
    attn_kernel<<<agrid, pblk, 0, stream>>>(Qb, Kb, Vb, AO);

    proj_gemm<float><<<pgrid, pblk, 0, stream>>>(AO, Wp, bp, (float*)d_out);
}
